// IndependentPolicy_77068893160318
// MI455X (gfx1250) — matrix operand order repaired and checked by reading, not compiled, not run on hardware
//
#include <hip/hip_runtime.h>
#include <math.h>

// MI455X / gfx1250: wave32, WMMA 16x16x32 f16 -> f32
typedef __attribute__((ext_vector_type(16))) _Float16 v16h;
typedef __attribute__((ext_vector_type(8)))  float    v8f;

#define HIDDEN 64
#define SLOTS  4
#define SEQ    24
#define VOCAB  64
#define ROWS   16     // batch rows per wave == WMMA M
#define WAVES  2      // waves per block, each owns an independent 16-row tile
#define LANES  32
#define SSTR   36     // stage row stride (dwords): 36k mod 64 spaced by 4 -> no bank conflicts

__device__ __forceinline__ v8f zero8() {
  v8f z;
#pragma unroll
  for (int i = 0; i < 8; ++i) z[i] = 0.0f;
  return z;
}

__device__ __forceinline__ v8f wmma_f16(v16h a, v16h b, v8f c) {
  // D = A(16x32 f16) * B(32x16 f16) + C(16x16 f32)
  return __builtin_amdgcn_wmma_f32_16x16x32_f16(false, a, false, b, (short)0, c,
                                                false, false);
}

// A-fragment (16x32 f16) from a row-major LDS tile.
// ISA 7.12.2: lanes 0-15 row M=lane hold K = k0+[0..7] (v0-3) and k0+16+[0..7]
// (v4-7); lanes 16-31 hold the +8 K offsets of the same rows.
__device__ __forceinline__ v16h mk_a(const _Float16* base, int stride, int lane, int k0) {
  const int row = lane & 15;
  const int kb  = k0 + ((lane >> 4) << 3);
  const _Float16* p = base + row * stride + kb;
  v16h a;
#pragma unroll
  for (int e = 0; e < 8; ++e) a[e] = p[e];
#pragma unroll
  for (int e = 0; e < 8; ++e) a[8 + e] = p[16 + e];
  return a;
}

// B-fragment (32x16 f16) from a row-major f32 global weight W[K][ncols]:
// lane = column N, K = k0 + 8*half + e (e<8), k0 + 16 + 8*half + (e-8) (e>=8).
__device__ __forceinline__ v16h mk_b(const float* W, int ncols, int k0, int n0, int lane) {
  const int col = n0 + (lane & 15);
  const int kb  = k0 + ((lane >> 4) << 3);
  v16h b;
#pragma unroll
  for (int e = 0; e < 16; ++e) b[e] = (_Float16)W[(kb + e + (e < 8 ? 0 : 8)) * ncols + col];
  return b;
}

__global__ __launch_bounds__(WAVES * LANES)
void slotmem_scan_kernel(const int* __restrict__ seqs,
                         const int* __restrict__ qtok,
                         const float* __restrict__ embed,
                         const float* __restrict__ wg_w1, const float* __restrict__ wg_b1,
                         const float* __restrict__ wg_w2, const float* __restrict__ wg_b2,
                         const float* __restrict__ eg_w1, const float* __restrict__ eg_b1,
                         const float* __restrict__ eg_w2, const float* __restrict__ eg_b2,
                         const float* __restrict__ rh_w1, const float* __restrict__ rh_b1,
                         const float* __restrict__ rh_w2, const float* __restrict__ rh_b2,
                         float* __restrict__ out)
{
  __shared__ _Float16 sNe[WAVES][ROWS * HIDDEN];           // token embedding tile (f16)
  __shared__ _Float16 sMem[WAVES][SLOTS * ROWS * HIDDEN];  // slot memory state (f16)
  // 5 staging regions (slots 0-3 = evict gate, region 4 = write gate), padded rows.
  // During readout the same buffer is aliased as the f16 [16][128] "r" tile.
  __shared__ float    sStage[WAVES][5 * ROWS * SSTR];
  __shared__ float    sW[WAVES][ROWS];                     // per-row write-gate sigmoid
  __shared__ int      sArg[WAVES][ROWS];                   // per-row evict argmax
  __shared__ float    sWgW2[32];
  __shared__ float    sEgW2[32];

  const int tid  = threadIdx.x;
  const int wv   = tid >> 5;
  const int ln   = tid & 31;
  const int row  = ln & 15;    // A-frag row AND C-frag column within an n-tile
  const int half = ln >> 4;
  const int b0   = (blockIdx.x * WAVES + wv) * ROWS;

  if (tid < 32)      sWgW2[tid]      = wg_w2[tid];
  else if (tid < 64) sEgW2[tid - 32] = eg_w2[tid - 32];

  // Persistent register B-fragments for the scan-phase weights.
  v16h wgB[2][2];                       // wg_w1[64,32]: 2 N-tiles x 2 K-tiles
#pragma unroll
  for (int n = 0; n < 2; ++n)
#pragma unroll
    for (int kt = 0; kt < 2; ++kt)
      wgB[n][kt] = mk_b(wg_w1, 32, 32 * kt, 16 * n, ln);

  v16h egB[2][4];                       // eg_w1[128,32]: rows 0-63 hit ne, 64-127 hit mem
#pragma unroll
  for (int n = 0; n < 2; ++n)
#pragma unroll
    for (int kt = 0; kt < 4; ++kt)
      egB[n][kt] = mk_b(eg_w1, 32, 32 * kt, 16 * n, ln);

  const float wgB1l[2] = { wg_b1[row], wg_b1[16 + row] };
  const float egB1l[2] = { eg_b1[row], eg_b1[16 + row] };
  const float wgB2v = wg_b2[0];
  const float egB2v = eg_b2[0];

  // mem0 = zeros
#pragma unroll 4
  for (int i = 0; i < SLOTS * ROWS * HIDDEN / LANES; ++i)
    sMem[wv][ln * (SLOTS * ROWS * HIDDEN / LANES) + i] = (_Float16)0.0f;
  __syncthreads();

  for (int t = 0; t < SEQ - 1; ++t) {
    // ---- gather token embedding -> sNe (lane: one row, 32 contiguous cols) ----
    {
      const int tok = seqs[(b0 + row) * SEQ + t];
      const float* er = embed + tok * HIDDEN + half * 32;
      _Float16* nr = &sNe[wv][row * HIDDEN + half * 32];
#pragma unroll 8
      for (int c = 0; c < 32; ++c) nr[c] = (_Float16)er[c];
    }
    __syncthreads();

    // ---- all WMMA work of this step under a single barrier section ----
    v16h aNe[2];                        // shared by write gate and evict-gate "ne" half
#pragma unroll
    for (int kt = 0; kt < 2; ++kt) aNe[kt] = mk_a(&sNe[wv][0], HIDDEN, ln, 32 * kt);

    // write gate: H1 = relu(ne @ wg_w1 + b1) -> stage region 4
#pragma unroll
    for (int n = 0; n < 2; ++n) {
      v8f c = zero8();
#pragma unroll
      for (int kt = 0; kt < 2; ++kt) c = wmma_f16(aNe[kt], wgB[n][kt], c);
      float* st = &sStage[wv][4 * ROWS * SSTR];
#pragma unroll
      for (int r = 0; r < 8; ++r)
        st[(half * 8 + r) * SSTR + n * 16 + row] = fmaxf(c[r] + wgB1l[n], 0.0f);
    }

    // evict gate: hoisted ne-half, then per-slot mem-half -> stage regions 0..3
    v8f cG[2];
#pragma unroll
    for (int n = 0; n < 2; ++n) {
      v8f c = zero8();
#pragma unroll
      for (int kt = 0; kt < 2; ++kt) c = wmma_f16(aNe[kt], egB[n][kt], c);
      cG[n] = c;
    }
    for (int s = 0; s < SLOTS; ++s) {
      v16h aM[2];
#pragma unroll
      for (int kt = 0; kt < 2; ++kt)
        aM[kt] = mk_a(&sMem[wv][s * ROWS * HIDDEN], HIDDEN, ln, 32 * kt);
      float* st = &sStage[wv][s * ROWS * SSTR];
#pragma unroll
      for (int n = 0; n < 2; ++n) {
        v8f c = cG[n];
#pragma unroll
        for (int kt = 0; kt < 2; ++kt) c = wmma_f16(aM[kt], egB[n][2 + kt], c);
#pragma unroll
        for (int r = 0; r < 8; ++r)
          st[(half * 8 + r) * SSTR + n * 16 + row] = fmaxf(c[r] + egB1l[n], 0.0f);
      }
    }
    __syncthreads();

    // ---- GEMV tails: lanes 0-15 evict argmax, lanes 16-31 write-gate sigmoid ----
    if (ln < 16) {
      float best = -3.4e38f;
      int   bests = 0;
      for (int s = 0; s < SLOTS; ++s) {
        const float* st = &sStage[wv][s * ROWS * SSTR + ln * SSTR];
        float acc = egB2v;
#pragma unroll 8
        for (int j = 0; j < 32; ++j) acc += st[j] * sEgW2[j];
        if (acc > best) { best = acc; bests = s; }  // strict '>' == first-max (argmax)
      }
      sArg[wv][ln] = bests;
    } else {
      const int r16 = ln - 16;
      const float* st = &sStage[wv][4 * ROWS * SSTR + r16 * SSTR];
      float acc = wgB2v;
#pragma unroll 8
      for (int j = 0; j < 32; ++j) acc += st[j] * sWgW2[j];
      sW[wv][r16] = 1.0f / (1.0f + expf(-acc));
    }
    __syncthreads();

    // ---- replace argmax slot with w * ne ----
    {
      const int   sa  = sArg[wv][row];
      const float wvv = sW[wv][row];
      const _Float16* nr = &sNe[wv][row * HIDDEN + half * 32];
      _Float16* mr = &sMem[wv][sa * ROWS * HIDDEN + row * HIDDEN + half * 32];
#pragma unroll 8
      for (int c = 0; c < 32; ++c) mr[c] = (_Float16)((float)nr[c] * wvv);
    }
    __syncthreads();
  }

  // ---- readout: r = cat(q, mean(mem)) [16,128], aliased into the stage buffer ----
  _Float16* sR = reinterpret_cast<_Float16*>(&sStage[wv][0]);
  {
    const int tok = qtok[b0 + row];
#pragma unroll 4
    for (int c = 0; c < HIDDEN; ++c) {
      const int col = half * HIDDEN + c;
      float v;
      if (col < HIDDEN) {
        v = embed[tok * HIDDEN + col];
      } else {
        const int h = col - HIDDEN;
        float m = 0.0f;
#pragma unroll
        for (int s = 0; s < SLOTS; ++s)
          m += (float)sMem[wv][s * ROWS * HIDDEN + row * HIDDEN + h];
        v = 0.25f * m;
      }
      sR[row * 2 * HIDDEN + col] = (_Float16)v;
    }
  }
  __syncthreads();

  // H = relu(r @ rh_w1 + b1) -> reuse sNe as f16 [16][64]
  {
    v16h aR[4];
#pragma unroll
    for (int kt = 0; kt < 4; ++kt) aR[kt] = mk_a(sR, 2 * HIDDEN, ln, 32 * kt);
#pragma unroll
    for (int n = 0; n < 4; ++n) {
      v8f c = zero8();
#pragma unroll
      for (int kt = 0; kt < 4; ++kt)
        c = wmma_f16(aR[kt], mk_b(rh_w1, HIDDEN, 32 * kt, 16 * n, ln), c);
      const float b1 = rh_b1[n * 16 + row];
#pragma unroll
      for (int r = 0; r < 8; ++r)
        sNe[wv][(half * 8 + r) * HIDDEN + n * 16 + row] = (_Float16)fmaxf(c[r] + b1, 0.0f);
    }
  }
  __syncthreads();

  // logits = H @ rh_w2 + b2 -> global (C-frag maps straight to [batch_row, vocab])
  {
    v16h aH[2];
#pragma unroll
    for (int kt = 0; kt < 2; ++kt) aH[kt] = mk_a(&sNe[wv][0], HIDDEN, ln, 32 * kt);
#pragma unroll
    for (int n = 0; n < 4; ++n) {
      v8f c = zero8();
#pragma unroll
      for (int kt = 0; kt < 2; ++kt)
        c = wmma_f16(aH[kt], mk_b(rh_w2, HIDDEN, 32 * kt, 16 * n, ln), c);
      const float b2 = rh_b2[n * 16 + row];
#pragma unroll
      for (int r = 0; r < 8; ++r)
        out[(b0 + half * 8 + r) * VOCAB + n * 16 + row] = c[r] + b2;
    }
  }
}

extern "C" void kernel_launch(void* const* d_in, const int* in_sizes, int n_in,
                              void* d_out, int out_size, void* d_ws, size_t ws_size,
                              hipStream_t stream) {
  const int*   seqs  = (const int*)d_in[0];
  const int*   qtok  = (const int*)d_in[1];
  const float* embed = (const float*)d_in[2];
  const float* wg_w1 = (const float*)d_in[3];
  const float* wg_b1 = (const float*)d_in[4];
  const float* wg_w2 = (const float*)d_in[5];
  const float* wg_b2 = (const float*)d_in[6];
  const float* eg_w1 = (const float*)d_in[7];
  const float* eg_b1 = (const float*)d_in[8];
  const float* eg_w2 = (const float*)d_in[9];
  const float* eg_b2 = (const float*)d_in[10];
  const float* rh_w1 = (const float*)d_in[11];
  const float* rh_b1 = (const float*)d_in[12];
  const float* rh_w2 = (const float*)d_in[13];
  const float* rh_b2 = (const float*)d_in[14];
  float* out = (float*)d_out;

  const int B = in_sizes[1];                 // query_tok has B elements
  const int blocks = B / (ROWS * WAVES);     // 16384 / 32 = 512 blocks of 64 threads
  slotmem_scan_kernel<<<blocks, WAVES * LANES, 0, stream>>>(
      seqs, qtok, embed,
      wg_w1, wg_b1, wg_w2, wg_b2,
      eg_w1, eg_b1, eg_w2, eg_b2,
      rh_w1, rh_b1, rh_w2, rh_b2, out);
}
